// SparseFeedForward_35897336660578
// MI455X (gfx1250) — matrix operand order repaired and checked by reading, not compiled, not run on hardware
//
#include <hip/hip_runtime.h>
#include <hip/hip_bf16.h>

typedef __bf16 bf16_t;
typedef __attribute__((ext_vector_type(16))) __bf16 v16bf;
typedef __attribute__((ext_vector_type(8)))  __bf16 v8bf;
typedef __attribute__((ext_vector_type(8)))  float  v8f;

#define D_DIM 1024
#define F_DIM 2048
#define E_NUM 8
#define N_TOK 4096
#define CAP   4096
#define A_STRIDE 1032   // 1024 + 8 bf16 pad -> rotates LDS banks, keeps 16B align
#define H_STRIDE 2056   // 2048 + 8 bf16 pad

union AFrag { v16bf v; v8bf h[2]; };
union BF4  { bf16_t b[4]; unsigned long long u; };

// ---------------------------------------------------------------------------
// 0. zero output + expert counts
// ---------------------------------------------------------------------------
__global__ void zero_kernel(float* __restrict__ out, int* __restrict__ counts, int n4) {
    const int i = blockIdx.x * blockDim.x + threadIdx.x;
    if (i < n4) ((float4*)out)[i] = make_float4(0.f, 0.f, 0.f, 0.f);
    if (blockIdx.x == 0 && threadIdx.x < E_NUM) counts[threadIdx.x] = 0;
}

// ---------------------------------------------------------------------------
// 1a. transpose last two dims + f32 -> bf16.  in: [E][R][C] f32, out: [E][C][R] bf16
//     block (32,8), grid (C/32, R/32, E)
// ---------------------------------------------------------------------------
__global__ void transpose_convert_kernel(const float* __restrict__ in,
                                         bf16_t* __restrict__ out,
                                         int R, int C) {
    __shared__ float tile[32][33];
    const int e  = blockIdx.z;
    const int c0 = blockIdx.x * 32;
    const int r0 = blockIdx.y * 32;
    const size_t base = (size_t)e * R * C;
    for (int i = threadIdx.y; i < 32; i += 8)
        tile[i][threadIdx.x] = in[base + (size_t)(r0 + i) * C + (c0 + threadIdx.x)];
    __syncthreads();
    for (int i = threadIdx.y; i < 32; i += 8)
        out[base + (size_t)(c0 + i) * R + (r0 + threadIdx.x)] =
            (bf16_t)tile[threadIdx.x][i];
}

// ---------------------------------------------------------------------------
// 1b. x: f32 -> bf16 (row-major, contiguous), 4 elements/thread
// ---------------------------------------------------------------------------
__global__ void convert_x_kernel(const float* __restrict__ x,
                                 bf16_t* __restrict__ xb, int n4) {
    const int i = blockIdx.x * blockDim.x + threadIdx.x;
    if (i >= n4) return;
    const float4 v = ((const float4*)x)[i];
    BF4 p;
    p.b[0] = (bf16_t)v.x; p.b[1] = (bf16_t)v.y;
    p.b[2] = (bf16_t)v.z; p.b[3] = (bf16_t)v.w;
    ((unsigned long long*)xb)[i] = p.u;
}

// ---------------------------------------------------------------------------
// 2. gating: one wave32 per token -> softmax, top-2, renorm, append to expert list
// ---------------------------------------------------------------------------
__global__ void gate_topk_kernel(const float* __restrict__ x,
                                 const float* __restrict__ gw,
                                 const float* __restrict__ gb,
                                 int* __restrict__ counts,
                                 int* __restrict__ tok_idx,
                                 float* __restrict__ tok_wt) {
    const int wave = threadIdx.x >> 5;
    const int lane = threadIdx.x & 31;
    const int t = blockIdx.x * 8 + wave;

    float acc[E_NUM];
#pragma unroll
    for (int e = 0; e < E_NUM; ++e) acc[e] = 0.f;

    const float* xr = x + (size_t)t * D_DIM;
    for (int d = lane; d < D_DIM; d += 32) {
        const float xv = xr[d];
        const float* g = gw + (size_t)d * E_NUM;
#pragma unroll
        for (int e = 0; e < E_NUM; ++e) acc[e] += xv * g[e];
    }
#pragma unroll
    for (int e = 0; e < E_NUM; ++e) {
#pragma unroll
        for (int off = 16; off > 0; off >>= 1)
            acc[e] += __shfl_xor(acc[e], off, 32);
    }

    if (lane == 0) {
        float logit[E_NUM];
        float m = -1e30f;
#pragma unroll
        for (int e = 0; e < E_NUM; ++e) {
            logit[e] = acc[e] + gb[e];
            m = fmaxf(m, logit[e]);
        }
        float p[E_NUM]; float s = 0.f;
#pragma unroll
        for (int e = 0; e < E_NUM; ++e) { p[e] = __expf(logit[e] - m); s += p[e]; }
        const float inv = 1.f / s;
        int i0 = 0; float v0 = -1.f;
#pragma unroll
        for (int e = 0; e < E_NUM; ++e) {
            const float pe = p[e] * inv;
            if (pe > v0) { v0 = pe; i0 = e; }
        }
        int i1 = 0; float v1 = -1.f;
#pragma unroll
        for (int e = 0; e < E_NUM; ++e) {
            if (e == i0) continue;
            const float pe = p[e] * inv;
            if (pe > v1) { v1 = pe; i1 = e; }
        }
        const float renorm = 1.f / (v0 + v1 + 1e-6f);
        const int pos0 = atomicAdd(&counts[i0], 1);
        tok_idx[i0 * CAP + pos0] = t;
        tok_wt [i0 * CAP + pos0] = v0 * renorm;
        const int pos1 = atomicAdd(&counts[i1], 1);
        tok_idx[i1 * CAP + pos1] = t;
        tok_wt [i1 * CAP + pos1] = v1 * renorm;
    }
}

// ---------------------------------------------------------------------------
// 3. fused grouped SwiGLU FFN. one block = (expert, 16-token tile), 8 waves.
//    x tile staged via CDNA5 async global->LDS DMA (no VGPR round trip).
// ---------------------------------------------------------------------------
__global__ void __launch_bounds__(256)
moe_ffn_kernel(const bf16_t* __restrict__ xb,
               const bf16_t* __restrict__ w1t, const float* __restrict__ b1,
               const bf16_t* __restrict__ w3t, const float* __restrict__ b3,
               const bf16_t* __restrict__ w2t, const float* __restrict__ b2,
               const int* __restrict__ counts, const int* __restrict__ tok_idx,
               const float* __restrict__ tok_wt, float* __restrict__ out) {
    __shared__ __align__(16) bf16_t sA[16 * A_STRIDE];  // gathered x tile, bf16
    __shared__ __align__(16) bf16_t sH[16 * H_STRIDE];  // SwiGLU hidden tile
    __shared__ int   sTok[16];
    __shared__ float sW[16];

    const int e    = blockIdx.x >> 8;     // 256 tiles per expert
    const int tile = blockIdx.x & 255;
    const int cnt  = counts[e];
    if (tile * 16 >= cnt) return;

    const int tid = threadIdx.x;
    if (tid < 16) {
        const int slot = tile * 16 + tid;
        if (slot < cnt) { sTok[tid] = tok_idx[e * CAP + slot]; sW[tid] = tok_wt[e * CAP + slot]; }
        else            { sTok[tid] = 0;                        sW[tid] = 0.f; }
    }
    __syncthreads();

    // ---- stage gathered bf16 x rows into LDS via async DMA ----
    // 16 rows x 2048B; each thread issues 8 x b128 (row = tid/16, 128B chunk = tid%16)
    {
        const int r = tid >> 4;
        const int c = (tid & 15) * 64;               // 64 bf16 = 128 B per thread
        const unsigned long long ga =
            (unsigned long long)(uintptr_t)(xb + (size_t)sTok[r] * D_DIM + c);
        const unsigned int lds =
            (unsigned int)(uintptr_t)&sA[r * A_STRIDE + c];
#pragma unroll
        for (int i = 0; i < 8; ++i) {
            asm volatile("global_load_async_to_lds_b128 %0, %1, off"
                         :: "v"(lds + i * 16), "v"(ga + (unsigned long long)(i * 16))
                         : "memory");
        }
    }
    asm volatile("s_wait_asynccnt 0" ::: "memory");
    __syncthreads();

    const int wave = tid >> 5;
    const int lane = tid & 31;
    const int col  = lane & 15;   // output column within 16x16 tile
    const int hi   = lane >> 4;   // half-wave select
    const int aoff = hi * 16;     // A-fragment K chunk start (elements)

    // ---- FFN1: h = silu(x@W1^T + b1) * (x@W3^T + b3), K = D ----
    for (int ft = wave; ft < F_DIM / 16; ft += 8) {
        const int f0 = ft * 16;
        v8f acc1 = {}; v8f acc3 = {};
        const bf16_t* pb1 = w1t + ((size_t)e * F_DIM + f0 + col) * D_DIM + hi * 16;
        const bf16_t* pb3 = w3t + ((size_t)e * F_DIM + f0 + col) * D_DIM + hi * 16;
#pragma unroll 4
        for (int k = 0; k < D_DIM / 32; ++k) {
            const int kb = k * 32;
            AFrag a;
            a.h[0] = *(const v8bf*)&sA[col * A_STRIDE + kb + aoff];
            a.h[1] = *(const v8bf*)&sA[col * A_STRIDE + kb + aoff + 8];
            const v16bf bv1 = *(const v16bf*)(pb1 + kb);
            const v16bf bv3 = *(const v16bf*)(pb3 + kb);
            __builtin_prefetch(pb1 + kb + 512, 0, 0);
            __builtin_prefetch(pb3 + kb + 512, 0, 0);
            acc1 = __builtin_amdgcn_wmma_f32_16x16x32_bf16(false, a.v, false, bv1,
                                                           (short)0, acc1, false, false);
            acc3 = __builtin_amdgcn_wmma_f32_16x16x32_bf16(false, a.v, false, bv3,
                                                           (short)0, acc3, false, false);
        }
        const float bias1 = b1[(size_t)e * F_DIM + f0 + col];
        const float bias3 = b3[(size_t)e * F_DIM + f0 + col];
#pragma unroll
        for (int j = 0; j < 8; ++j) {
            const int m = j + hi * 8;                 // token row in tile
            const float u  = acc1[j] + bias1;
            const float sg = u / (1.f + __expf(-u));  // silu
            const float hv = sg * (acc3[j] + bias3);
            sH[m * H_STRIDE + f0 + col] = (bf16_t)hv;
        }
    }
    __syncthreads();

    // ---- FFN2: y = h @ W2^T + b2, scale by gate weight, atomic combine ----
    for (int dt = wave; dt < D_DIM / 16; dt += 8) {
        const int d0 = dt * 16;
        v8f acc = {};
        const bf16_t* pb = w2t + ((size_t)e * D_DIM + d0 + col) * F_DIM + hi * 16;
#pragma unroll 4
        for (int k = 0; k < F_DIM / 32; ++k) {
            const int kb = k * 32;
            AFrag a;
            a.h[0] = *(const v8bf*)&sH[col * H_STRIDE + kb + aoff];
            a.h[1] = *(const v8bf*)&sH[col * H_STRIDE + kb + aoff + 8];
            const v16bf bv = *(const v16bf*)(pb + kb);
            __builtin_prefetch(pb + kb + 512, 0, 0);
            acc = __builtin_amdgcn_wmma_f32_16x16x32_bf16(false, a.v, false, bv,
                                                          (short)0, acc, false, false);
        }
        const float bias2 = b2[(size_t)e * D_DIM + d0 + col];
#pragma unroll
        for (int j = 0; j < 8; ++j) {
            const int m = j + hi * 8;
            const float val = (acc[j] + bias2) * sW[m];
            unsafeAtomicAdd(&out[(size_t)sTok[m] * D_DIM + d0 + col], val);
        }
    }
}

// ---------------------------------------------------------------------------
// host launcher
// ---------------------------------------------------------------------------
extern "C" void kernel_launch(void* const* d_in, const int* in_sizes, int n_in,
                              void* d_out, int out_size, void* d_ws, size_t ws_size,
                              hipStream_t stream) {
    (void)in_sizes; (void)n_in; (void)out_size; (void)ws_size;
    const float* x      = (const float*)d_in[0];
    const float* w1     = (const float*)d_in[1];
    const float* b1     = (const float*)d_in[2];
    const float* w3     = (const float*)d_in[3];
    const float* b3     = (const float*)d_in[4];
    const float* w2     = (const float*)d_in[5];
    const float* b2     = (const float*)d_in[6];
    const float* gate_w = (const float*)d_in[7];
    const float* gate_b = (const float*)d_in[8];
    float* out = (float*)d_out;

    const size_t wbytes = (size_t)E_NUM * D_DIM * F_DIM * sizeof(bf16_t); // 32 MiB each
    char* ws = (char*)d_ws;
    bf16_t* w1t     = (bf16_t*)(ws);
    bf16_t* w3t     = (bf16_t*)(ws + wbytes);
    bf16_t* w2t     = (bf16_t*)(ws + 2 * wbytes);
    bf16_t* xb      = (bf16_t*)(ws + 3 * wbytes);                 // 8 MiB
    char*   tail    = ws + 3 * wbytes + (size_t)N_TOK * D_DIM * sizeof(bf16_t);
    int*    tok_idx = (int*)  (tail);
    float*  tok_wt  = (float*)(tail + (size_t)E_NUM * CAP * 4);
    int*    counts  = (int*)  (tail + 2 * (size_t)E_NUM * CAP * 4);

    const int n4 = N_TOK * D_DIM / 4;
    zero_kernel<<<(n4 + 255) / 256, 256, 0, stream>>>(out, counts, n4);

    const dim3 tb(32, 8);
    transpose_convert_kernel<<<dim3(F_DIM/32, D_DIM/32, E_NUM), tb, 0, stream>>>(w1, w1t, D_DIM, F_DIM);
    transpose_convert_kernel<<<dim3(F_DIM/32, D_DIM/32, E_NUM), tb, 0, stream>>>(w3, w3t, D_DIM, F_DIM);
    transpose_convert_kernel<<<dim3(D_DIM/32, F_DIM/32, E_NUM), tb, 0, stream>>>(w2, w2t, F_DIM, D_DIM);
    convert_x_kernel<<<(n4 + 255) / 256, 256, 0, stream>>>(x, xb, n4);

    gate_topk_kernel<<<N_TOK / 8, 256, 0, stream>>>(x, gate_w, gate_b, counts, tok_idx, tok_wt);

    moe_ffn_kernel<<<E_NUM * (CAP / 16), 256, 0, stream>>>(
        xb, w1t, b1, w3t, b3, w2t, b2, counts, tok_idx, tok_wt, out);
}
